// GraphAttentionLayer_53154515255605
// MI455X (gfx1250) — matrix operand order repaired and checked by reading, not compiled, not run on hardware
//
#include <hip/hip_runtime.h>
#include <math.h>

// Problem constants (from reference)
#define NN   30000
#define EE   480000
#define CC   128
#define HH   8
#define DHH  16
#define NBB  32
#define NAA  16
#define FCC  64
#define AHH  128

typedef __attribute__((ext_vector_type(16))) __bf16 v16bf;
typedef __attribute__((ext_vector_type(8)))  __bf16 v8bf;
typedef __attribute__((ext_vector_type(4)))  __bf16 v4bf;
typedef __attribute__((ext_vector_type(8)))  float  v8f;
typedef __attribute__((ext_vector_type(4)))  int    v4i;

#define AS1 __attribute__((address_space(1)))
#define AS3 __attribute__((address_space(3)))

#if defined(__has_builtin)
#if __has_builtin(__builtin_amdgcn_global_load_async_to_lds_b128) && \
    __has_builtin(__builtin_amdgcn_s_wait_asynccnt)
#define HAVE_ASYNC 1
#endif
#endif
#ifndef HAVE_ASYNC
#define HAVE_ASYNC 0
#endif

static __device__ inline v8f wmma_bf16(v16bf a, v16bf b, v8f c) {
  return __builtin_amdgcn_wmma_f32_16x16x32_bf16(false, a, false, b, (short)0, c,
                                                 false, false);
}

// A fragment (16x32) from row-major *bf16* tile [16][ldk]: two ds_load_b128.
// lane<16: M=lane, K=(j&7)+((j&8)?16:0) ; lane>=16: +8 on K, M=lane-16.
static __device__ inline v16bf a_frag_bf(const __bf16* __restrict__ tile, int ldk,
                                         int kb, int lane) {
  const int m   = lane & 15;
  const int off = (lane & 16) ? 8 : 0;
  const __bf16* row = tile + m * ldk + kb + off;
  v8bf lo = *(const v8bf*)(row);
  v8bf hi = *(const v8bf*)(row + 16);
  return __builtin_shufflevector(lo, hi, 0, 1, 2, 3, 4, 5, 6, 7,
                                 8, 9, 10, 11, 12, 13, 14, 15);
}

// A fragment from row-major *f32* memory (global or LDS): 4x b128 + cvt.
static __device__ inline v16bf a_frag_f32(const float* __restrict__ tile, int ldk,
                                          int kb, int lane) {
  const int m   = lane & 15;
  const int off = (lane & 16) ? 8 : 0;
  const float* row = tile + (size_t)m * ldk + kb + off;
  const float4 x0 = *(const float4*)(row + 0);
  const float4 x1 = *(const float4*)(row + 4);
  const float4 x2 = *(const float4*)(row + 16);
  const float4 x3 = *(const float4*)(row + 20);
  v16bf a;
  a[0] = (__bf16)x0.x;  a[1] = (__bf16)x0.y;  a[2] = (__bf16)x0.z;  a[3] = (__bf16)x0.w;
  a[4] = (__bf16)x1.x;  a[5] = (__bf16)x1.y;  a[6] = (__bf16)x1.z;  a[7] = (__bf16)x1.w;
  a[8] = (__bf16)x2.x;  a[9] = (__bf16)x2.y;  a[10] = (__bf16)x2.z; a[11] = (__bf16)x2.w;
  a[12] = (__bf16)x3.x; a[13] = (__bf16)x3.y; a[14] = (__bf16)x3.z; a[15] = (__bf16)x3.w;
  return a;
}

// B fragment (32x16) from packed bf16 weights Wb[N][K] (k-contiguous per n):
// two global_load_b128, no conversion.
static __device__ inline v16bf b_frag_p(const __bf16* __restrict__ Wb, int K,
                                        int kb, int n0, int lane) {
  const int n  = n0 + (lane & 15);
  const int k0 = kb + ((lane & 16) ? 8 : 0);
  const __bf16* p = Wb + (size_t)n * K + k0;
  v8bf lo = *(const v8bf*)(p);
  v8bf hi = *(const v8bf*)(p + 16);
  return __builtin_shufflevector(lo, hi, 0, 1, 2, 3, 4, 5, 6, 7,
                                 8, 9, 10, 11, 12, 13, 14, 15);
}

static __device__ inline float silu_f(float x) { return x / (1.f + expf(-x)); }

static __device__ inline float slrelu_f(float x) {
  float s = 1.f / (1.f + expf(-x));
  return 0.6f * x + 0.4f * x * (2.f * s - 1.f);
}

static __device__ inline void atomicMaxFloat(float* addr, float val) {
  if (val >= 0.f) atomicMax((int*)addr, __float_as_int(val));
  else            atomicMin((unsigned int*)addr, __float_as_uint(val));
}

// ------------------------------------------------ weight pack (f32 -> bf16^T)
__global__ __launch_bounds__(256) void pack_w_kernel(const float* __restrict__ in,
                                                     __bf16* __restrict__ out,
                                                     int K, int N) {
  const int id = blockIdx.x * 256 + threadIdx.x;
  if (id >= K * N) return;
  const int n = id / K, k = id - n * K;
  out[id] = (__bf16)in[(size_t)k * N + n];   // out[n][k] = in[k][n]
}

// ---------------------------------------------------------------- init
__global__ __launch_bounds__(256) void init_kernel(float* __restrict__ m,
                                                   float* __restrict__ den,
                                                   float* __restrict__ agg) {
  int id = blockIdx.x * 256 + threadIdx.x;
  if (id < NN * HH) { m[id] = -__builtin_inff(); den[id] = 0.f; }
  if (id < NN * AHH) agg[id] = 0.f;
}

// ------------------------------------------------- node projections P/Q
__global__ __launch_bounds__(256) void node_proj_kernel(
    const float* __restrict__ X, const __bf16* __restrict__ Wb_src,
    const float* __restrict__ bsrc, const __bf16* __restrict__ Wb_dst,
    float* __restrict__ P, float* __restrict__ Q) {
  const int wid  = (blockIdx.x * 256 + threadIdx.x) >> 5;
  const int lane = threadIdx.x & 31;
  const int row0 = wid * 16;
  if (row0 >= NN) return;
  const int nl = lane & 15, moff = (lane & 16) ? 8 : 0;

  v16bf af[4];
#pragma unroll
  for (int kb = 0; kb < 4; ++kb)
    af[kb] = a_frag_f32(X + (size_t)row0 * CC, CC, kb * 32, lane);

  for (int nt = 0; nt < 8; ++nt) {
    const int n0 = nt * 16;
    v8f cp = {}, cq = {};
#pragma unroll
    for (int kb = 0; kb < 4; ++kb) {
      cp = wmma_bf16(af[kb], b_frag_p(Wb_src, CC, kb * 32, n0, lane), cp);
      cq = wmma_bf16(af[kb], b_frag_p(Wb_dst, CC, kb * 32, n0, lane), cq);
    }
    const float bias = bsrc[n0 + nl];
#pragma unroll
    for (int r = 0; r < 8; ++r) {
      const size_t o = (size_t)(row0 + moff + r) * CC + n0 + nl;
      P[o] = cp[r] + bias;
      Q[o] = cq[r];
    }
  }
}

// ---------------------------------------------------- fused edge kernel
// Per-wave LDS slice layout (bytes):
//   s_wf  f32  [16][64]  @ 0      (4096)
//   s_red f32  [16][16]  @ 4096   (1024)
//   s_h1  bf16 [16][64]  @ 5120   (2048)
//   s_h2  bf16 [16][64]  @ 7168   (2048)
//   s_msg bf16 [16][128] @ 9216   (4096)
//   s_w   bf16 [16][128] @ 13312  (4096)  -> total 17408 B/wave
#define EDGE_LDS_BYTES 17408

__global__ __launch_bounds__(64) void edge_kernel(
    const float* __restrict__ P, const float* __restrict__ Q,
    const float* __restrict__ node_attr, const float* __restrict__ edge_attr,
    const float* __restrict__ edge_emb,
    const __bf16* __restrict__ Wb_fc1, const float* __restrict__ b_fc1,
    const __bf16* __restrict__ Wb_fc2, const float* __restrict__ b_fc2,
    const __bf16* __restrict__ Wb_fc3,
    const __bf16* __restrict__ Wb_alpha, const __bf16* __restrict__ Wb_lin,
    const float* __restrict__ w_int, const __bf16* __restrict__ Wb_val,
    const float* __restrict__ att_dot,
    const int* __restrict__ edge_src, const int* __restrict__ edge_dst,
    float* __restrict__ logits, float* __restrict__ mbuf,
    __bf16* __restrict__ vbuf) {
  __shared__ __align__(16) char smem[2][EDGE_LDS_BYTES];
  const int wslot = threadIdx.x >> 5;
  const int lane  = threadIdx.x & 31;
  const int e0    = (blockIdx.x * 2 + wslot) * 16;
  if (e0 >= EE) return;                               // wave-uniform
  char* base = smem[wslot];
  float*  s_wf  = (float*)(base);
  float*  s_red = (float*)(base + 4096);
  __bf16* s_h1  = (__bf16*)(base + 5120);
  __bf16* s_h2  = (__bf16*)(base + 7168);
  __bf16* s_msg = (__bf16*)(base + 9216);
  __bf16* s_w   = (__bf16*)(base + 13312);
  const int nl = lane & 15, moff = (lane & 16) ? 8 : 0;

  __builtin_prefetch(edge_emb + (size_t)(e0 + 256) * NBB, 0, 1);

  // ---- stage wf = [edge_emb | attr[src] | attr[dst]]  (16x64 f32) into LDS
#if HAVE_ASYNC
  {
    AS3 char* sw3 = (AS3 char*)s_wf;
    const char* ge = (const char*)edge_emb + (size_t)e0 * NBB * 4;
#pragma unroll
    for (int t = 0; t < 4; ++t) {        // 16 rows x 128B of edge_emb
      const int c = lane + 32 * t;       // 16B chunk id (128 total)
      const int m = c >> 3, q = c & 7;
      __builtin_amdgcn_global_load_async_to_lds_b128(
          (AS1 v4i*)(ge + c * 16), (AS3 v4i*)(sw3 + m * 256 + q * 16), 0, 0);
    }
#pragma unroll
    for (int t = 0; t < 2; ++t) {        // 16 rows x 64B gathers (src & dst)
      const int c = lane + 32 * t;       // 64 chunks of 16B
      const int m = c >> 2, q = c & 3;
      const char* gs = (const char*)node_attr +
                       (size_t)edge_src[e0 + m] * (NAA * 4);
      const char* gd = (const char*)node_attr +
                       (size_t)edge_dst[e0 + m] * (NAA * 4);
      __builtin_amdgcn_global_load_async_to_lds_b128(
          (AS1 v4i*)(gs + q * 16), (AS3 v4i*)(sw3 + m * 256 + 128 + q * 16), 0, 0);
      __builtin_amdgcn_global_load_async_to_lds_b128(
          (AS1 v4i*)(gd + q * 16), (AS3 v4i*)(sw3 + m * 256 + 192 + q * 16), 0, 0);
    }
    __builtin_amdgcn_s_wait_asynccnt(0);
    asm volatile("" ::: "memory");
  }
#else
  for (int idx = lane; idx < 16 * 64; idx += 32) {
    const int m = idx >> 6, c = idx & 63;
    const int e = e0 + m;
    float v;
    if (c < NBB)            v = edge_emb[(size_t)e * NBB + c];
    else if (c < NBB + NAA) v = node_attr[(size_t)edge_src[e] * NAA + (c - NBB)];
    else                    v = node_attr[(size_t)edge_dst[e] * NAA + (c - NBB - NAA)];
    s_wf[idx] = v;
  }
#endif
  __builtin_amdgcn_wave_barrier();

  // ---- h1 = silu(wf @ W_fc1 + b1)  -> s_h1 (bf16)
  for (int nt = 0; nt < 4; ++nt) {
    const int n0 = nt * 16;
    v8f c = {};
#pragma unroll
    for (int kb = 0; kb < 2; ++kb)
      c = wmma_bf16(a_frag_f32(s_wf, 64, kb * 32, lane),
                    b_frag_p(Wb_fc1, FCC, kb * 32, n0, lane), c);
    const float bias = b_fc1[n0 + nl];
#pragma unroll
    for (int r = 0; r < 8; ++r)
      s_h1[(moff + r) * 64 + n0 + nl] = (__bf16)silu_f(c[r] + bias);
  }
  __builtin_amdgcn_wave_barrier();

  // ---- h2 = silu(h1 @ W_fc2 + b2)  -> s_h2 (bf16)
  for (int nt = 0; nt < 4; ++nt) {
    const int n0 = nt * 16;
    v8f c = {};
#pragma unroll
    for (int kb = 0; kb < 2; ++kb)
      c = wmma_bf16(a_frag_bf(s_h1, 64, kb * 32, lane),
                    b_frag_p(Wb_fc2, FCC, kb * 32, n0, lane), c);
    const float bias = b_fc2[n0 + nl];
#pragma unroll
    for (int r = 0; r < 8; ++r)
      s_h2[(moff + r) * 64 + n0 + nl] = (__bf16)silu_f(c[r] + bias);
  }
  __builtin_amdgcn_wave_barrier();

  // ---- weight = h2 @ W_fc3  -> s_w (bf16, 16x128)
  for (int nt = 0; nt < 8; ++nt) {
    const int n0 = nt * 16;
    v8f c = {};
#pragma unroll
    for (int kb = 0; kb < 2; ++kb)
      c = wmma_bf16(a_frag_bf(s_h2, 64, kb * 32, lane),
                    b_frag_p(Wb_fc3, FCC, kb * 32, n0, lane), c);
#pragma unroll
    for (int r = 0; r < 8; ++r)
      s_w[(moff + r) * CC + n0 + nl] = (__bf16)c[r];
  }
  __builtin_amdgcn_wave_barrier();

  // ---- message = (P[src]+Q[dst]) * edge_attr * weight  -> s_msg (bf16)
  // vectorized: 4 channels / lane-iteration
  for (int c = lane; c < 512; c += 32) {
    const int m = c >> 5, k4 = (c & 31) * 4;
    const int e = e0 + m;
    const int s = edge_src[e], d = edge_dst[e];
    const float ea = edge_attr[e];
    const float4 p = *(const float4*)(P + (size_t)s * CC + k4);
    const float4 q = *(const float4*)(Q + (size_t)d * CC + k4);
    const v4bf  w = *(const v4bf*)(s_w + m * CC + k4);
    v4bf o;
    o[0] = (__bf16)((p.x + q.x) * ea * (float)w[0]);
    o[1] = (__bf16)((p.y + q.y) * ea * (float)w[1]);
    o[2] = (__bf16)((p.z + q.z) * ea * (float)w[2]);
    o[3] = (__bf16)((p.w + q.w) * ea * (float)w[3]);
    *(v4bf*)(s_msg + m * CC + k4) = o;
  }
  __builtin_amdgcn_wave_barrier();

  // ---- scores = msg @ W_alpha ; smooth-leaky ; logits + segment max
  for (int head = 0; head < HH; ++head) {
    const int n0 = head * 16;
    v8f c = {};
#pragma unroll
    for (int kb = 0; kb < 4; ++kb)
      c = wmma_bf16(a_frag_bf(s_msg, CC, kb * 32, lane),
                    b_frag_p(Wb_alpha, CC, kb * 32, n0, lane), c);
    const float ad = att_dot[head * DHH + nl];
#pragma unroll
    for (int r = 0; r < 8; ++r)
      s_red[(moff + r) * 16 + nl] = slrelu_f(c[r]) * ad;
    __builtin_amdgcn_wave_barrier();
    if (lane < 16) {
      float lg = 0.f;
#pragma unroll
      for (int d = 0; d < 16; ++d) lg += s_red[lane * 16 + d];
      const int e = e0 + lane;
      logits[(size_t)e * HH + head] = lg;
      atomicMaxFloat(mbuf + (size_t)edge_dst[e] * HH + head, lg);
    }
    __builtin_amdgcn_wave_barrier();
  }

  // ---- v1 = silu(msg @ W_lin)  -> s_w (bf16)
  for (int nt = 0; nt < 8; ++nt) {
    const int n0 = nt * 16;
    v8f c = {};
#pragma unroll
    for (int kb = 0; kb < 4; ++kb)
      c = wmma_bf16(a_frag_bf(s_msg, CC, kb * 32, lane),
                    b_frag_p(Wb_lin, CC, kb * 32, n0, lane), c);
#pragma unroll
    for (int r = 0; r < 8; ++r)
      s_w[(moff + r) * CC + n0 + nl] = (__bf16)silu_f(c[r]);
  }
  __builtin_amdgcn_wave_barrier();

  // ---- v2 = v1 * edge_attr * w_int  -> s_msg (bf16), 8 channels / iteration
  for (int c = lane; c < 256; c += 32) {
    const int m = c >> 4, k8 = (c & 15) * 8;
    const float ea = edge_attr[e0 + m];
    const v8bf  v  = *(const v8bf*)(s_w + m * CC + k8);
    const float4 w0 = *(const float4*)(w_int + k8);
    const float4 w1 = *(const float4*)(w_int + k8 + 4);
    v8bf o;
    o[0] = (__bf16)((float)v[0] * ea * w0.x);
    o[1] = (__bf16)((float)v[1] * ea * w0.y);
    o[2] = (__bf16)((float)v[2] * ea * w0.z);
    o[3] = (__bf16)((float)v[3] * ea * w0.w);
    o[4] = (__bf16)((float)v[4] * ea * w1.x);
    o[5] = (__bf16)((float)v[5] * ea * w1.y);
    o[6] = (__bf16)((float)v[6] * ea * w1.z);
    o[7] = (__bf16)((float)v[7] * ea * w1.w);
    *(v8bf*)(s_msg + m * CC + k8) = o;
  }
  __builtin_amdgcn_wave_barrier();

  // ---- value = v2 @ W_val  -> vbuf (bf16)
  for (int nt = 0; nt < 8; ++nt) {
    const int n0 = nt * 16;
    v8f c = {};
#pragma unroll
    for (int kb = 0; kb < 4; ++kb)
      c = wmma_bf16(a_frag_bf(s_msg, CC, kb * 32, lane),
                    b_frag_p(Wb_val, CC, kb * 32, n0, lane), c);
#pragma unroll
    for (int r = 0; r < 8; ++r)
      vbuf[(size_t)(e0 + moff + r) * AHH + n0 + nl] = (__bf16)c[r];
  }
}

// -------------------------------------------- exp + segment denominator
__global__ __launch_bounds__(256) void softmax_den_kernel(
    float* __restrict__ logits, const float* __restrict__ mbuf,
    float* __restrict__ den, const int* __restrict__ edge_dst) {
  const int id = blockIdx.x * 256 + threadIdx.x;
  if (id >= EE * HH) return;
  const int e = id >> 3, h = id & 7;
  const int d = edge_dst[e];
  float mm = mbuf[(size_t)d * HH + h];
  if (!(mm > -1e30f)) mm = 0.f;
  const float ex = expf(logits[id] - mm);
  logits[id] = ex;
  atomicAdd(den + (size_t)d * HH + h, ex);
}

// --------------------------------- attention-weighted scatter aggregate
__global__ __launch_bounds__(256) void aggregate_kernel(
    const float* __restrict__ exbuf, const float* __restrict__ den,
    const __bf16* __restrict__ vbuf, const int* __restrict__ edge_dst,
    float* __restrict__ agg) {
  const int id = blockIdx.x * 256 + threadIdx.x;   // E*16 threads, 8 elems each
  if (id >= EE * 16) return;
  const int e = id >> 4, c0 = (id & 15) * 8, h = c0 >> 4;
  const int d = edge_dst[e];
  const float alpha = exbuf[(size_t)e * HH + h] /
                      (den[(size_t)d * HH + h] + 1e-16f);
  const v8bf v = *(const v8bf*)(vbuf + (size_t)e * AHH + c0);
  float* dst = agg + (size_t)d * AHH + c0;
#pragma unroll
  for (int j = 0; j < 8; ++j)
    atomicAdd(dst + j, (float)v[j] * alpha);
}

// ------------------------------------------------------ output proj
__global__ __launch_bounds__(256) void out_proj_kernel(
    const float* __restrict__ agg, const __bf16* __restrict__ Wb_out,
    const float* __restrict__ b_out, float* __restrict__ out) {
  const int wid  = (blockIdx.x * 256 + threadIdx.x) >> 5;
  const int lane = threadIdx.x & 31;
  const int row0 = wid * 16;
  if (row0 >= NN) return;
  const int nl = lane & 15, moff = (lane & 16) ? 8 : 0;

  v16bf af[4];
#pragma unroll
  for (int kb = 0; kb < 4; ++kb)
    af[kb] = a_frag_f32(agg + (size_t)row0 * AHH, AHH, kb * 32, lane);

  for (int nt = 0; nt < 8; ++nt) {
    const int n0 = nt * 16;
    v8f c = {};
#pragma unroll
    for (int kb = 0; kb < 4; ++kb)
      c = wmma_bf16(af[kb], b_frag_p(Wb_out, AHH, kb * 32, n0, lane), c);
    const float bias = b_out[n0 + nl];
#pragma unroll
    for (int r = 0; r < 8; ++r)
      out[(size_t)(row0 + moff + r) * CC + n0 + nl] = c[r] + bias;
  }
}

// ------------------------------------------------------------ launcher
extern "C" void kernel_launch(void* const* d_in, const int* in_sizes, int n_in,
                              void* d_out, int out_size, void* d_ws, size_t ws_size,
                              hipStream_t stream) {
  const float* node_feats = (const float*)d_in[0];
  const float* node_attr  = (const float*)d_in[1];
  const float* edge_attr  = (const float*)d_in[2];
  const float* edge_emb   = (const float*)d_in[3];
  const float* W_src      = (const float*)d_in[4];
  const float* b_src      = (const float*)d_in[5];
  const float* W_dst      = (const float*)d_in[6];
  const float* W_fc1      = (const float*)d_in[7];
  const float* b_fc1      = (const float*)d_in[8];
  const float* W_fc2      = (const float*)d_in[9];
  const float* b_fc2      = (const float*)d_in[10];
  const float* W_fc3      = (const float*)d_in[11];
  const float* W_alpha    = (const float*)d_in[12];
  const float* W_lin      = (const float*)d_in[13];
  const float* w_int      = (const float*)d_in[14];
  const float* W_val      = (const float*)d_in[15];
  const float* att_dot    = (const float*)d_in[16];
  const float* W_out      = (const float*)d_in[17];
  const float* b_out      = (const float*)d_in[18];
  const int*   edge_src   = (const int*)d_in[19];
  const int*   edge_dst   = (const int*)d_in[20];

  char* ws = (char*)d_ws;
  float* P      = (float*)ws;  ws += (size_t)NN * CC  * sizeof(float);
  float* Q      = (float*)ws;  ws += (size_t)NN * CC  * sizeof(float);
  float* logits = (float*)ws;  ws += (size_t)EE * HH  * sizeof(float);
  float* mbuf   = (float*)ws;  ws += (size_t)NN * HH  * sizeof(float);
  float* den    = (float*)ws;  ws += (size_t)NN * HH  * sizeof(float);
  float* agg    = (float*)ws;  ws += (size_t)NN * AHH * sizeof(float);
  __bf16* vbuf  = (__bf16*)ws; ws += (size_t)EE * AHH * sizeof(__bf16);
  // packed bf16 weights (N-major, k-contiguous)
  __bf16* Wb_src   = (__bf16*)ws; ws += (size_t)CC  * CC  * 2;
  __bf16* Wb_dst   = (__bf16*)ws; ws += (size_t)CC  * CC  * 2;
  __bf16* Wb_fc1   = (__bf16*)ws; ws += (size_t)FCC * FCC * 2;
  __bf16* Wb_fc2   = (__bf16*)ws; ws += (size_t)FCC * FCC * 2;
  __bf16* Wb_fc3   = (__bf16*)ws; ws += (size_t)FCC * CC  * 2;
  __bf16* Wb_alpha = (__bf16*)ws; ws += (size_t)CC  * AHH * 2;
  __bf16* Wb_lin   = (__bf16*)ws; ws += (size_t)CC  * CC  * 2;
  __bf16* Wb_val   = (__bf16*)ws; ws += (size_t)CC  * AHH * 2;
  __bf16* Wb_out   = (__bf16*)ws;

  // ---- pack weights to bf16 (transposed)
  pack_w_kernel<<<(CC * CC + 255) / 256, 256, 0, stream>>>(W_src, Wb_src, CC, CC);
  pack_w_kernel<<<(CC * CC + 255) / 256, 256, 0, stream>>>(W_dst, Wb_dst, CC, CC);
  pack_w_kernel<<<(FCC * FCC + 255) / 256, 256, 0, stream>>>(W_fc1, Wb_fc1, FCC, FCC);
  pack_w_kernel<<<(FCC * FCC + 255) / 256, 256, 0, stream>>>(W_fc2, Wb_fc2, FCC, FCC);
  pack_w_kernel<<<(FCC * CC + 255) / 256, 256, 0, stream>>>(W_fc3, Wb_fc3, FCC, CC);
  pack_w_kernel<<<(CC * AHH + 255) / 256, 256, 0, stream>>>(W_alpha, Wb_alpha, CC, AHH);
  pack_w_kernel<<<(CC * CC + 255) / 256, 256, 0, stream>>>(W_lin, Wb_lin, CC, CC);
  pack_w_kernel<<<(CC * AHH + 255) / 256, 256, 0, stream>>>(W_val, Wb_val, CC, AHH);
  pack_w_kernel<<<(AHH * CC + 255) / 256, 256, 0, stream>>>(W_out, Wb_out, AHH, CC);

  init_kernel<<<(NN * AHH + 255) / 256, 256, 0, stream>>>(mbuf, den, agg);

  const int node_waves  = NN / 16;               // 1875
  const int node_blocks = (node_waves + 7) / 8;  // 235
  node_proj_kernel<<<node_blocks, 256, 0, stream>>>(node_feats, Wb_src, b_src,
                                                    Wb_dst, P, Q);

  edge_kernel<<<EE / 32, 64, 0, stream>>>(P, Q, node_attr, edge_attr, edge_emb,
                                          Wb_fc1, b_fc1, Wb_fc2, b_fc2, Wb_fc3,
                                          Wb_alpha, Wb_lin, w_int, Wb_val, att_dot,
                                          edge_src, edge_dst, logits, mbuf, vbuf);

  softmax_den_kernel<<<(EE * HH + 255) / 256, 256, 0, stream>>>(logits, mbuf,
                                                                den, edge_dst);

  aggregate_kernel<<<(EE * 16 + 255) / 256, 256, 0, stream>>>(logits, den, vbuf,
                                                              edge_dst, agg);

  out_proj_kernel<<<node_blocks, 256, 0, stream>>>(agg, Wb_out, b_out,
                                                   (float*)d_out);
}
